// SpanEncoder_36756330119608
// MI455X (gfx1250) — matrix operand order repaired and checked by reading, not compiled, not run on hardware
//
#include <hip/hip_runtime.h>
#include <hip/hip_bf16.h>

#define B_    64
#define S_    512
#define D_    256
#define H_    256
#define G4H   1024
#define F_    768      // 2H + D
#define LMAXC 64
#define OUTIN 3584     // 2F + HEADS*F + 2H

typedef __attribute__((ext_vector_type(16))) _Float16 v16h;
typedef __attribute__((ext_vector_type(8)))  _Float16 v8h;
typedef __attribute__((ext_vector_type(8)))  float    v8f;

// ---------------- WMMA fragment loaders (contiguous k order, k = 16*(lane/16) + e, same for A and B) ----------------
// A: 16x32 f16 from row-major MxK source, leading dim `ld` (multiple of 8).
// lane<16: M=lane, e0..15 = K 0..15 ; lane>=16: K 16..31.
__device__ __forceinline__ v16h load_a16(const _Float16* base, int ld) {
  const int lane = threadIdx.x & 31;
  const int m = lane & 15, hf = lane >> 4;
  v8h lo = *(const v8h*)(base + (size_t)m * ld + hf * 16);
  v8h hi = *(const v8h*)(base + (size_t)m * ld + hf * 16 + 8);
  v16h r;
#pragma unroll
  for (int i = 0; i < 8; ++i) { r[i] = lo[i]; r[i + 8] = hi[i]; }
  return r;
}

// B: 32x16 f16 taken from an [N][K] row-major weight matrix (ld = K).
// lane<16: N=lane, K=e ; lane>=16: N=lane-16, K=16+e  -> 32 contiguous bytes.
__device__ __forceinline__ v16h load_b16(const _Float16* base, int ld) {
  const int lane = threadIdx.x & 31;
  const int n = lane & 15, hf = lane >> 4;
  const v8h* p = (const v8h*)(base + (size_t)n * ld + hf * 16);
  v8h lo = p[0], hi = p[1];
  v16h r;
#pragma unroll
  for (int i = 0; i < 8; ++i) { r[i] = lo[i]; r[i + 8] = hi[i]; }
  return r;
}

__device__ __forceinline__ float sigf(float x)  { return 1.f / (1.f + __expf(-x)); }
__device__ __forceinline__ float lrelu(float x) { return x > 0.f ? x : 0.01f * x; }

// ---------------- fp32 -> f16 conversion ----------------
__global__ void cvt_f16(const float* __restrict__ in, _Float16* __restrict__ out, int n) {
  int i = blockIdx.x * 256 + threadIdx.x;
  if (i < n) out[i] = (_Float16)in[i];
}

// ---------------- Input-projection GEMM with transposed fp32 output -----------
// A: [B][S][K] f16, W: [1024][K] f16, outT: [S][1024][64] f32 (gate preacts,
// biases folded). One wave = one time step t, all 4 batch tiles x 2 N-tiles.
// reverse==1: store at the padded-reversed time slot (per-row seq length).
template <int K>
__global__ __launch_bounds__(256) void gemm_proj_t(
    const _Float16* __restrict__ A, const _Float16* __restrict__ W,
    const float* __restrict__ b1, const float* __restrict__ b2,
    const int* __restrict__ lens, int reverse,
    float* __restrict__ outT) {
  const int wv = (blockIdx.x * blockDim.x + threadIdx.x) >> 5;
  const int t  = wv >> 5;          // time step 0..511
  const int ng = wv & 31;          // N group (2 tiles of 16)
  if (t >= S_) return;
  const int n0 = ng * 32;
  const _Float16* Abase = A + (size_t)t * K;   // element (bm,k) at bm*S*K + k

  v8f acc[4][2] = {};
#pragma unroll
  for (int k = 0; k < K; k += 32) {            // fully unrolled: loads can hoist
    v16h a0 = load_a16(Abase + (size_t)0 * 16 * S_ * K + k, S_ * K);
    v16h a1 = load_a16(Abase + (size_t)1 * 16 * S_ * K + k, S_ * K);
    v16h a2 = load_a16(Abase + (size_t)2 * 16 * S_ * K + k, S_ * K);
    v16h a3 = load_a16(Abase + (size_t)3 * 16 * S_ * K + k, S_ * K);
    v16h w0 = load_b16(W + (size_t)(n0)      * K + k, K);
    v16h w1 = load_b16(W + (size_t)(n0 + 16) * K + k, K);
    acc[0][0] = __builtin_amdgcn_wmma_f32_16x16x32_f16(false, a0, false, w0, (short)0, acc[0][0], false, false);
    acc[1][0] = __builtin_amdgcn_wmma_f32_16x16x32_f16(false, a1, false, w0, (short)0, acc[1][0], false, false);
    acc[2][0] = __builtin_amdgcn_wmma_f32_16x16x32_f16(false, a2, false, w0, (short)0, acc[2][0], false, false);
    acc[3][0] = __builtin_amdgcn_wmma_f32_16x16x32_f16(false, a3, false, w0, (short)0, acc[3][0], false, false);
    acc[0][1] = __builtin_amdgcn_wmma_f32_16x16x32_f16(false, a0, false, w1, (short)0, acc[0][1], false, false);
    acc[1][1] = __builtin_amdgcn_wmma_f32_16x16x32_f16(false, a1, false, w1, (short)0, acc[1][1], false, false);
    acc[2][1] = __builtin_amdgcn_wmma_f32_16x16x32_f16(false, a2, false, w1, (short)0, acc[2][1], false, false);
    acc[3][1] = __builtin_amdgcn_wmma_f32_16x16x32_f16(false, a3, false, w1, (short)0, acc[3][1], false, false);
  }

  const int lane = threadIdx.x & 31;
  const int nc = lane & 15, hf = lane >> 4;
  if (!reverse) {
#pragma unroll
    for (int mt = 0; mt < 4; ++mt)
#pragma unroll
      for (int j = 0; j < 2; ++j) {
        int n = n0 + j * 16 + nc;
        float bias = b1[n] + b2[n];
        v8f vv;
#pragma unroll
        for (int r = 0; r < 8; ++r) vv[r] = acc[mt][j][r] + bias;
        *(v8f*)(outT + ((size_t)t * G4H + n) * 64 + mt * 16 + hf * 8) = vv;
      }
  } else {
    int otv[4][8];
#pragma unroll
    for (int mt = 0; mt < 4; ++mt)
#pragma unroll
      for (int r = 0; r < 8; ++r) {
        int bm = mt * 16 + hf * 8 + r;
        int len = lens[bm];
        otv[mt][r] = (t < len) ? (len - 1 - t) : t;
      }
#pragma unroll
    for (int mt = 0; mt < 4; ++mt)
#pragma unroll
      for (int j = 0; j < 2; ++j) {
        int n = n0 + j * 16 + nc;
        float bias = b1[n] + b2[n];
#pragma unroll
        for (int r = 0; r < 8; ++r)
          outT[((size_t)otv[mt][r] * G4H + n) * 64 + mt * 16 + hf * 8 + r] =
              acc[mt][j][r] + bias;
      }
  }
}

// ---------------- Recurrent scan: 4 workgroups = (dir, batch-half) -------------
// Batch rows are independent in the recurrence, so each WG owns 32 rows and its
// own 16KB h-state; only the hidden-unit dim stays within a workgroup.
// xpT: [2][S][1024][64] f32 (sequential in t for BOTH dirs; bwd pre-reversed)
// whh: [2][1024][256] f16 ; hcat: [B][S][512] f16 ; hn: [2][B][256] f32
__global__ __launch_bounds__(512, 1) void lstm_scan(
    const float* __restrict__ xpT, const _Float16* __restrict__ whh,
    const int* __restrict__ lens, _Float16* __restrict__ hcat,
    float* __restrict__ hn) {
  const int dir  = blockIdx.x >> 1;     // 0 = forward, 1 = backward
  const int mh   = blockIdx.x & 1;      // batch half: rows mh*32 .. mh*32+31
  const int wave = threadIdx.x >> 5;    // 0..15 : hidden-unit group
  const int lane = threadIdx.x & 31;
  const int nc = lane & 15, hf = lane >> 4;
  const int ncol = wave * 16 + nc;      // hidden unit 0..255

  __shared__ __align__(16) _Float16 hbuf[32 * 256];   // h state (local rows)
  __shared__ int lens_s[32];
  for (int i = threadIdx.x; i < 32 * 256; i += 512) hbuf[i] = (_Float16)0.f;
  if (threadIdx.x < 32) lens_s[threadIdx.x] = lens[mh * 32 + threadIdx.x];
  __syncthreads();

  const float*    xpd = xpT + (size_t)dir * S_ * G4H * 64 + mh * 32;
  const _Float16* Wd  = whh + (size_t)dir * G4H * H_;

  float cst[2][8];
#pragma unroll
  for (int mt = 0; mt < 2; ++mt)
#pragma unroll
    for (int r = 0; r < 8; ++r) cst[mt][r] = 0.f;

  for (int t = 0; t < S_; ++t) {
    // gate preactivations: straight v8f load into the accumulators (no cvt)
    v8f acc[4][2];                       // [gate][local batch-tile]
#pragma unroll
    for (int g = 0; g < 4; ++g)
#pragma unroll
      for (int mt = 0; mt < 2; ++mt)
        acc[g][mt] = *(const v8f*)(xpd + ((size_t)t * G4H + g * 256 + ncol) * 64 +
                                   mt * 16 + hf * 8);
    // g += h @ whh^T. K-slice order rotated by t so the weight-fragment loads
    // are NOT loop-invariant: stops LICM from hoisting 256 VGPRs of B-frags out
    // of the 512-step loop (which caused scratch spills); the sum over K slices
    // is order-independent.
#pragma unroll
    for (int kk = 0; kk < 8; ++kk) {
      const int kp = (kk + t) & 7;
      v16h a0 = load_a16(hbuf + 0 * 16 * 256 + kp * 32, 256);
      v16h a1 = load_a16(hbuf + 1 * 16 * 256 + kp * 32, 256);
#pragma unroll
      for (int g = 0; g < 4; ++g) {
        v16h b = load_b16(Wd + (size_t)(g * 256 + wave * 16) * 256 + kp * 32, 256);
        acc[g][0] = __builtin_amdgcn_wmma_f32_16x16x32_f16(false, a0, false, b, (short)0, acc[g][0], false, false);
        acc[g][1] = __builtin_amdgcn_wmma_f32_16x16x32_f16(false, a1, false, b, (short)0, acc[g][1], false, false);
      }
    }
    __syncthreads();                     // all waves done reading hbuf
#pragma unroll
    for (int mt = 0; mt < 2; ++mt)
#pragma unroll
      for (int r = 0; r < 8; ++r) {
        float iv = sigf(acc[0][mt][r]);
        float fv = sigf(acc[1][mt][r]);
        float gv = tanhf(acc[2][mt][r]);
        float ov = sigf(acc[3][mt][r]);
        float c  = fv * cst[mt][r] + iv * gv;
        cst[mt][r] = c;
        float h = ov * tanhf(c);
        int lm  = mt * 16 + hf * 8 + r;          // local row 0..31
        int bm  = mh * 32 + lm;                  // global batch row
        hbuf[lm * 256 + ncol] = (_Float16)h;
        int len   = lens_s[lm];
        bool valid = t < len;
        int ot = (dir == 1) ? (valid ? (len - 1 - t) : t) : t;   // un-reverse
        hcat[((size_t)bm * S_ + ot) * 512 + dir * 256 + ncol] = (_Float16)(valid ? h : 0.f);
        if (t == len - 1) hn[((size_t)dir * B_ + bm) * H_ + ncol] = h;
      }
    __syncthreads();                     // hbuf updated before next step
  }
}

// ---------------- Span attention + output MLP + LayerNorm ----------------------
__global__ __launch_bounds__(256, 1) void span_final(
    const float* __restrict__ words, const _Float16* __restrict__ words_h,
    const _Float16* __restrict__ hcat2, const float* __restrict__ hn,
    const int* __restrict__ spans, _Float16* __restrict__ cont_g,
    const _Float16* __restrict__ attw_h, const float* __restrict__ att_ih_b,
    const float* __restrict__ att_ho_w, const float* __restrict__ att_ho_b,
    const float* __restrict__ out_w1, const float* __restrict__ out_b1,
    const float* __restrict__ out_w2, const float* __restrict__ out_b2,
    const float* __restrict__ ln_g, const float* __restrict__ ln_b,
    float* __restrict__ out) {
  const int b = blockIdx.x, tid = threadIdx.x;
  const int s0 = spans[b * 2 + 0];
  const int e  = spans[b * 2 + 1];
  const int L  = e - s0;

  __shared__ __align__(16) _Float16 sff[64 * 256];  // 32 KB
  __shared__ float enc[OUTIN];                      // 14 KB
  __shared__ float wsm[128], rawm[128];
  __shared__ float h12[512];

  _Float16* cont = cont_g + (size_t)b * LMAXC * F_;

  // contents: rows start..start+63, clipped, masked
  for (int idx = tid; idx < LMAXC * F_; idx += 256) {
    int i = idx / F_, c = idx - i * F_;
    _Float16 v = (_Float16)0.f;
    if (i < L) {
      int s = s0 + i; if (s > S_ - 1) s = S_ - 1;
      v = (c < 512) ? hcat2[((size_t)b * S_ + s) * 512 + c]
                    : words_h[((size_t)b * S_ + s) * 256 + (c - 512)];
    }
    cont[idx] = v;
  }
  __syncthreads();   // also fences block-local global writes

  // ff = leaky(contents @ att_ih_w^T + b)   M=64, N=256, K=768 via WMMA
  {
    const int wv = tid >> 5, lane = tid & 31;
    const int mt = wv & 3, nh = wv >> 2;
    const int nc = lane & 15, hf = lane >> 4;
    for (int jn = 0; jn < 8; ++jn) {
      int n0 = (nh * 8 + jn) * 16;
      v8f acc = {};
#pragma unroll
      for (int k = 0; k < F_; k += 32) {
        v16h a  = load_a16(cont + (size_t)mt * 16 * F_ + k, F_);
        v16h bb = load_b16(attw_h + (size_t)n0 * F_ + k, F_);
        acc = __builtin_amdgcn_wmma_f32_16x16x32_f16(false, a, false, bb,
                                                     (short)0, acc, false, false);
      }
      int n = n0 + nc;
      float bias = att_ih_b[n];
#pragma unroll
      for (int r = 0; r < 8; ++r)
        sff[(mt * 16 + hf * 8 + r) * 256 + n] = (_Float16)lrelu(acc[r] + bias);
    }
  }
  // enc[0:1536) recurrent = tok[s0] ++ tok[e-1]
  for (int j = tid; j < 1536; j += 256) {
    int s = (j < F_) ? s0 : (e - 1);
    int c = (j < F_) ? j : j - F_;
    enc[j] = (c < 512) ? (float)hcat2[((size_t)b * S_ + s) * 512 + c]
                       : words[((size_t)b * S_ + s) * 256 + (c - 512)];
  }
  // enc[3072:3584) context = hn_f ++ hn_b
  enc[3072 + tid] = hn[(size_t)0 * B_ * 256 + b * 256 + tid];
  enc[3328 + tid] = hn[(size_t)1 * B_ * 256 + b * 256 + tid];
  __syncthreads();

  // raw = ff @ att_ho_w^T + b, masked
  if (tid < 128) {
    int i = tid >> 1, k = tid & 1;
    float a = att_ho_b[k];
    for (int j = 0; j < 256; ++j) a += (float)sff[i * 256 + j] * att_ho_w[k * 256 + j];
    rawm[i * 2 + k] = (i < L) ? a : -1e32f;
  }
  __syncthreads();
  // softmax over span positions, per head
  if (tid < 2) {
    int k = tid;
    float mx = rawm[k];
    for (int i = 1; i < 64; ++i) mx = fmaxf(mx, rawm[i * 2 + k]);
    float s = 0.f;
    for (int i = 0; i < 64; ++i) s += __expf(rawm[i * 2 + k] - mx);
    float inv = 1.f / s;
    for (int i = 0; i < 64; ++i) wsm[i * 2 + k] = __expf(rawm[i * 2 + k] - mx) * inv;
  }
  __syncthreads();
  // attended -> enc[1536:3072), reshape (F, HEADS) row-major
  for (int j = tid; j < F_; j += 256) {
    float a0 = 0.f, a1 = 0.f;
    for (int i = 0; i < 64; ++i) {
      float cv = (float)cont[i * F_ + j];
      a0 += cv * wsm[i * 2 + 0];
      a1 += cv * wsm[i * 2 + 1];
    }
    enc[1536 + j * 2 + 0] = a0;
    enc[1536 + j * 2 + 1] = a1;
  }
  __syncthreads();
  // h1 = leaky(enc @ out_w1^T + b1)
  {
    float a = out_b1[tid];
    const float* wr = out_w1 + (size_t)tid * OUTIN;
    for (int j = 0; j < OUTIN; ++j) a += enc[j] * wr[j];
    h12[tid] = lrelu(a);
  }
  __syncthreads();
  // h2 = leaky(h1 @ out_w2^T + b2)
  {
    float a = out_b2[tid];
    const float* wr = out_w2 + (size_t)tid * 256;
    for (int j = 0; j < 256; ++j) a += h12[j] * wr[j];
    h12[256 + tid] = lrelu(a);
  }
  __syncthreads();
  // LayerNorm
  {
    float mu = 0.f;
    for (int j = 0; j < 256; ++j) mu += h12[256 + j];
    mu *= (1.f / 256.f);
    float var = 0.f;
    for (int j = 0; j < 256; ++j) { float d = h12[256 + j] - mu; var += d * d; }
    var *= (1.f / 256.f);
    out[(size_t)b * 256 + tid] =
        (h12[256 + tid] - mu) * rsqrtf(var + 1e-5f) * ln_g[tid] + ln_b[tid];
  }
}

// ---------------- host launcher -------------------------------------------------
extern "C" void kernel_launch(void* const* d_in, const int* in_sizes, int n_in,
                              void* d_out, int out_size, void* d_ws, size_t ws_size,
                              hipStream_t stream) {
  (void)in_sizes; (void)n_in; (void)out_size; (void)ws_size;
  const float* words    = (const float*)d_in[0];
  const int*   seq_lens = (const int*)d_in[1];
  const int*   spans    = (const int*)d_in[2];
  const float* wih[4] = { (const float*)d_in[3],  (const float*)d_in[7],
                          (const float*)d_in[11], (const float*)d_in[15] };
  const float* whhw[4] = { (const float*)d_in[4],  (const float*)d_in[8],
                           (const float*)d_in[12], (const float*)d_in[16] };
  const float* bih[4] = { (const float*)d_in[5],  (const float*)d_in[9],
                          (const float*)d_in[13], (const float*)d_in[17] };
  const float* bhh[4] = { (const float*)d_in[6],  (const float*)d_in[10],
                          (const float*)d_in[14], (const float*)d_in[18] };
  const float* att_ih_w = (const float*)d_in[19];
  const float* att_ih_b = (const float*)d_in[20];
  const float* att_ho_w = (const float*)d_in[21];
  const float* att_ho_b = (const float*)d_in[22];
  const float* out_w1 = (const float*)d_in[23];
  const float* out_b1 = (const float*)d_in[24];
  const float* out_w2 = (const float*)d_in[25];
  const float* out_b2 = (const float*)d_in[26];
  const float* ln_g = (const float*)d_in[27];
  const float* ln_b = (const float*)d_in[28];
  float* out = (float*)d_out;

  char* ws = (char*)d_ws;
  size_t off = 0;
  auto take = [&](size_t bytes) {
    char* p = ws + off;
    off = (off + bytes + 255) & ~(size_t)255;
    return p;
  };
  _Float16* words_h = (_Float16*)take((size_t)B_ * S_ * D_ * 2);
  _Float16* wih0_h  = (_Float16*)take((size_t)2 * G4H * D_ * 2);
  _Float16* wih1_h  = (_Float16*)take((size_t)2 * G4H * 512 * 2);
  _Float16* whh0_h  = (_Float16*)take((size_t)2 * G4H * H_ * 2);
  _Float16* whh1_h  = (_Float16*)take((size_t)2 * G4H * H_ * 2);
  _Float16* attw_h  = (_Float16*)take((size_t)256 * F_ * 2);
  float*    xpT     = (float*)take((size_t)2 * S_ * G4H * 64 * 4);
  _Float16* hcat1   = (_Float16*)take((size_t)B_ * S_ * 512 * 2);
  _Float16* hcat2   = (_Float16*)take((size_t)B_ * S_ * 512 * 2);
  _Float16* cont_g  = (_Float16*)take((size_t)B_ * LMAXC * F_ * 2);
  float*    hn      = (float*)take((size_t)2 * B_ * H_ * 4);

  auto cvt = [&](const float* s, _Float16* d, int n) {
    cvt_f16<<<(n + 255) / 256, 256, 0, stream>>>(s, d, n);
  };
  cvt(words, words_h, B_ * S_ * D_);
  cvt(wih[0], wih0_h,             G4H * D_);
  cvt(wih[1], wih0_h + G4H * D_,  G4H * D_);
  cvt(wih[2], wih1_h,             G4H * 512);
  cvt(wih[3], wih1_h + G4H * 512, G4H * 512);
  cvt(whhw[0], whh0_h,             G4H * H_);
  cvt(whhw[1], whh0_h + G4H * H_,  G4H * H_);
  cvt(whhw[2], whh1_h,             G4H * H_);
  cvt(whhw[3], whh1_h + G4H * H_,  G4H * H_);
  cvt(att_ih_w, attw_h, 256 * F_);

  const size_t xpStride = (size_t)S_ * G4H * 64;
  const int projBlocks = (S_ * 32) / 8;   // 16384 waves / 8 per block = 2048

  // layer 0 input projections (biases folded: bih + bhh); bwd pre-reversed
  gemm_proj_t<256><<<projBlocks, 256, 0, stream>>>(
      words_h, wih0_h, bih[0], bhh[0], seq_lens, 0, xpT);
  gemm_proj_t<256><<<projBlocks, 256, 0, stream>>>(
      words_h, wih0_h + G4H * D_, bih[1], bhh[1], seq_lens, 1, xpT + xpStride);
  // layer 0 recurrent scan (dir x batch-half)
  lstm_scan<<<4, 512, 0, stream>>>(xpT, whh0_h, seq_lens, hcat1, hn);

  // layer 1 input projections on concat(out_f, out_b)
  gemm_proj_t<512><<<projBlocks, 256, 0, stream>>>(
      hcat1, wih1_h, bih[2], bhh[2], seq_lens, 0, xpT);
  gemm_proj_t<512><<<projBlocks, 256, 0, stream>>>(
      hcat1, wih1_h + G4H * 512, bih[3], bhh[3], seq_lens, 1, xpT + xpStride);
  // layer 1 recurrent scan
  lstm_scan<<<4, 512, 0, stream>>>(xpT, whh1_h, seq_lens, hcat2, hn);

  // span attention + MLP + LayerNorm
  span_final<<<B_, 256, 0, stream>>>(words, words_h, hcat2, hn, spans, cont_g,
                                     attw_h, att_ih_b, att_ho_w, att_ho_b,
                                     out_w1, out_b1, out_w2, out_b2,
                                     ln_g, ln_b, out);
}
